// MultiHeadSelfAttention_3289944949321
// MI455X (gfx1250) — matrix operand order repaired and checked by reading, not compiled, not run on hardware
//
#include <hip/hip_runtime.h>
#include <hip/hip_bf16.h>

typedef __attribute__((ext_vector_type(16))) __bf16 v16bf;
typedef __attribute__((ext_vector_type(8)))  __bf16 v8bf;
typedef __attribute__((ext_vector_type(8)))  float  v8f;
typedef __attribute__((ext_vector_type(4)))  int    v4i;

#define ATT_B 2
#define ATT_S 2048
#define ATT_E 1024
#define ATT_H 16
#define ATT_D 64

#define GLOBAL_AS __attribute__((address_space(1)))
#define LDS_AS    __attribute__((address_space(3)))

#if defined(__has_builtin)
#if __has_builtin(__builtin_amdgcn_global_load_async_to_lds_b128)
#define HAVE_ASYNC_LDS 1
#endif
#endif

#if defined(HAVE_ASYNC_LDS)
// signature (probe-confirmed by clang diagnostic): (v4i AS1*, v4i AS3*, imm int, imm int)
#define ASYNC_B128(gsrc, ldst) \
    __builtin_amdgcn_global_load_async_to_lds_b128( \
        (GLOBAL_AS v4i*)(v4i*)(gsrc), (LDS_AS v4i*)(v4i*)(ldst), 0, 0)
#if __has_builtin(__builtin_amdgcn_s_wait_asynccnt)
#define WAIT_ASYNC0() __builtin_amdgcn_s_wait_asynccnt(0)
#else
#define WAIT_ASYNC0() asm volatile("s_wait_asynccnt 0" ::: "memory")
#endif
#endif

// ---------------------------------------------------------------------------
// f32 -> bf16 elementwise convert
// ---------------------------------------------------------------------------
__global__ void k_f32_to_bf16(const float* __restrict__ in, __bf16* __restrict__ out, int n) {
    int i = blockIdx.x * blockDim.x + threadIdx.x;
    if (i < n) out[i] = (__bf16)in[i];
}

// ---------------------------------------------------------------------------
// W[K,N] f32 -> Wt[N,K] bf16 (tiled transpose via LDS)
// ---------------------------------------------------------------------------
__global__ __launch_bounds__(256)
void k_transpose_bf16(const float* __restrict__ W, __bf16* __restrict__ Wt, int rows, int cols) {
    __shared__ float tile[32][33];
    int x = blockIdx.x * 32 + threadIdx.x;
    for (int j = threadIdx.y; j < 32; j += 8) {
        int y = blockIdx.y * 32 + j;
        tile[j][threadIdx.x] = W[(size_t)y * cols + x];
    }
    __syncthreads();
    for (int j = threadIdx.y; j < 32; j += 8) {
        int n = blockIdx.x * 32 + j;
        int k = blockIdx.y * 32 + threadIdx.x;
        Wt[(size_t)n * rows + k] = (__bf16)tile[threadIdx.x][j];
    }
}

// ---------------------------------------------------------------------------
// C[M,N] f32 = A[M,K] bf16 * Bt[N,K]^T bf16 + bias[N]
// Block: 256 threads = 8 waves; block tile 128(M) x 128(N); K-chunk 32.
// Waves laid out 4(M) x 2(N): wave tile 32x64 = 2x4 WMMA 16x16x32 tiles.
// Double-buffered LDS staging via GLOBAL_LOAD_ASYNC_TO_LDS_B128 (ASYNCcnt).
// ---------------------------------------------------------------------------
#define GK_BM 128
#define GK_BN 128
#define GK_BK 32

__device__ __forceinline__
void gk_stage(const __bf16* __restrict__ A, const __bf16* __restrict__ Bt,
              int K, int m_blk, int n_blk, int k0,
              __bf16 (*sA)[GK_BK], __bf16 (*sB)[GK_BK], int tid) {
#if defined(HAVE_ASYNC_LDS)
    #pragma unroll
    for (int p = 0; p < 2; ++p) {                 // A: 128x32 = 512 16B chunks
        int idx = tid + p * 256;
        int r   = idx >> 2;
        int ck  = (idx & 3) * 8;
        ASYNC_B128(&A[(size_t)(m_blk + r) * K + k0 + ck], &sA[r][ck]);
    }
    #pragma unroll
    for (int p = 0; p < 2; ++p) {                 // B: 128x32
        int idx = tid + p * 256;
        int r   = idx >> 2;
        int ck  = (idx & 3) * 8;
        ASYNC_B128(&Bt[(size_t)(n_blk + r) * K + k0 + ck], &sB[r][ck]);
    }
#else
    #pragma unroll
    for (int p = 0; p < 2; ++p) {
        int idx = tid + p * 256;
        int r   = idx >> 2;
        int ck  = (idx & 3) * 8;
        *(v8bf*)&sA[r][ck] = *(const v8bf*)&A[(size_t)(m_blk + r) * K + k0 + ck];
    }
    #pragma unroll
    for (int p = 0; p < 2; ++p) {
        int idx = tid + p * 256;
        int r   = idx >> 2;
        int ck  = (idx & 3) * 8;
        *(v8bf*)&sB[r][ck] = *(const v8bf*)&Bt[(size_t)(n_blk + r) * K + k0 + ck];
    }
#endif
}

__global__ __launch_bounds__(256)
void k_gemm_bf16_wmma(const __bf16* __restrict__ A, const __bf16* __restrict__ Bt,
                      const float* __restrict__ bias, float* __restrict__ C,
                      int M, int N, int K) {
    __shared__ __align__(32) __bf16 sA[2][GK_BM][GK_BK];
    __shared__ __align__(32) __bf16 sB[2][GK_BN][GK_BK];
    const int tid  = threadIdx.x;
    const int lane = tid & 31;
    const int w    = tid >> 5;
    const int wm   = w >> 1;          // 0..3 -> 32 rows each
    const int wn   = w & 1;           // 0..1 -> 64 cols each
    const int l15  = lane & 15;
    const int lhi  = lane >> 4;
    const int klo  = lhi * 8;
    const int m_blk = blockIdx.y * GK_BM;
    const int n_blk = blockIdx.x * GK_BN;
    const int NT = K / GK_BK;

    v8f c[2][4] = {};
    union U { v16bf v; v8bf h8[2]; };

#if defined(HAVE_ASYNC_LDS)
    gk_stage(A, Bt, K, m_blk, n_blk, 0, sA[0], sB[0], tid);
    for (int kt = 0; kt < NT; ++kt) {
        WAIT_ASYNC0();                 // this wave's async tile landed in LDS
        __syncthreads();               // all waves' tiles landed; prev compute done
        if (kt + 1 < NT)
            gk_stage(A, Bt, K, m_blk, n_blk, (kt + 1) * GK_BK,
                     sA[(kt + 1) & 1], sB[(kt + 1) & 1], tid);
        const int buf = kt & 1;
#else
    for (int kt = 0; kt < NT; ++kt) {
        const int buf = 0;
        gk_stage(A, Bt, K, m_blk, n_blk, kt * GK_BK, sA[0], sB[0], tid);
        __syncthreads();
#endif
        v16bf a[2];
        #pragma unroll
        for (int sm = 0; sm < 2; ++sm) {
            int row = wm * 32 + sm * 16 + l15;
            U u;
            u.h8[0] = *(const v8bf*)&sA[buf][row][klo];
            u.h8[1] = *(const v8bf*)&sA[buf][row][16 + klo];
            a[sm] = u.v;
        }
        #pragma unroll
        for (int sn = 0; sn < 4; ++sn) {
            int row = wn * 64 + sn * 16 + l15;
            U ub;
            ub.h8[0] = *(const v8bf*)&sB[buf][row][klo];
            ub.h8[1] = *(const v8bf*)&sB[buf][row][16 + klo];
            v16bf b = ub.v;
            #pragma unroll
            for (int sm = 0; sm < 2; ++sm)
                c[sm][sn] = __builtin_amdgcn_wmma_f32_16x16x32_bf16(
                    false, a[sm], false, b, (short)0, c[sm][sn], false, false);
        }
#if !defined(HAVE_ASYNC_LDS)
        __syncthreads();
#endif
    }

    #pragma unroll
    for (int sm = 0; sm < 2; ++sm) {
        #pragma unroll
        for (int sn = 0; sn < 4; ++sn) {
            int col = n_blk + wn * 64 + sn * 16 + l15;
            float bc = bias ? bias[col] : 0.0f;
            #pragma unroll
            for (int r = 0; r < 8; ++r) {
                int rowm = m_blk + wm * 32 + sm * 16 + r + lhi * 8;
                C[(size_t)rowm * N + col] = c[sm][sn][r] + bc;
            }
        }
    }
}

// ---------------------------------------------------------------------------
// RoPE: Qf/Kf f32 [B,S,H,D] -> Qr/Kr bf16 [B,H,S,D]; Q pre-scaled by 1/sqrt(D)
// ---------------------------------------------------------------------------
__global__ void k_rope_pack(const float* __restrict__ Qf, const float* __restrict__ Kf,
                            const int* __restrict__ pos,
                            __bf16* __restrict__ Qr, __bf16* __restrict__ Kr) {
    int idx = blockIdx.x * blockDim.x + threadIdx.x;
    int i = idx & 31;
    int s = (idx >> 5) & (ATT_S - 1);
    int h = (idx >> 16) & (ATT_H - 1);
    int b = (idx >> 20) & (ATT_B - 1);

    float p    = (float)pos[b * ATT_S + s];
    float freq = __powf(10000.0f, -(float)i * (1.0f / 32.0f));
    float ang  = p * freq;
    float sn, cs;
    __sincosf(ang, &sn, &cs);

    size_t src = ((size_t)(b * ATT_S + s) * ATT_H + h) * ATT_D + 2 * i;
    size_t dst = ((size_t)(b * ATT_H + h) * ATT_S + s) * ATT_D + 2 * i;

    float q0 = Qf[src], q1 = Qf[src + 1];
    Qr[dst]     = (__bf16)((q0 * cs - q1 * sn) * 0.125f);
    Qr[dst + 1] = (__bf16)((q0 * sn + q1 * cs) * 0.125f);

    float k0 = Kf[src], k1 = Kf[src + 1];
    Kr[dst]     = (__bf16)(k0 * cs - k1 * sn);
    Kr[dst + 1] = (__bf16)(k0 * sn + k1 * cs);
}

// ---------------------------------------------------------------------------
// V f32 [B,S,H,D] -> Vt bf16 [B,H,D,S]
// ---------------------------------------------------------------------------
__global__ void k_v_transpose(const float* __restrict__ Vf, __bf16* __restrict__ Vt) {
    int idx = blockIdx.x * blockDim.x + threadIdx.x;
    int s = idx & (ATT_S - 1);
    int d = (idx >> 11) & (ATT_D - 1);
    int h = (idx >> 17) & (ATT_H - 1);
    int b = (idx >> 21) & (ATT_B - 1);
    Vt[idx] = (__bf16)Vf[((size_t)(b * ATT_S + s) * ATT_H + h) * ATT_D + d];
}

// ---------------------------------------------------------------------------
// Flash attention: one wave per 16 queries; 8 waves/block -> 128 q per block.
// grid = (S/128, B*H). Streams 32 keys/iter with online softmax.
// ---------------------------------------------------------------------------
__global__ __launch_bounds__(256)
void k_attn_fa(const __bf16* __restrict__ Qr, const __bf16* __restrict__ Kr,
               const __bf16* __restrict__ Vt, __bf16* __restrict__ Ctx) {
    __shared__ __align__(32) __bf16 sP[8][16 * 32];
    const int lane = threadIdx.x & 31;
    const int w    = threadIdx.x >> 5;
    const int l15  = lane & 15;
    const int lhi  = lane >> 4;
    const int klo  = lhi * 8;
    const int bh   = blockIdx.y;
    const int b    = bh >> 4;
    const int h    = bh & 15;
    const int q0   = blockIdx.x * 128 + w * 16;

    const size_t qkBase = (size_t)bh * ATT_S * ATT_D;
    const size_t vBase  = (size_t)bh * ATT_D * ATT_S;

    union U { v16bf v; v8bf h8[2]; };

    v16bf aq[2];
    {
        const __bf16* qrow = Qr + qkBase + (size_t)(q0 + l15) * ATT_D;
        #pragma unroll
        for (int j = 0; j < 2; ++j) {
            U u;
            u.h8[0] = *(const v8bf*)(qrow + j * 32 + klo);
            u.h8[1] = *(const v8bf*)(qrow + j * 32 + 16 + klo);
            aq[j] = u.v;
        }
    }

    float m_i[8], l_i[8];
    #pragma unroll
    for (int r = 0; r < 8; ++r) { m_i[r] = -1e30f; l_i[r] = 0.0f; }
    v8f acc[4] = {};

    __bf16* myP = &sP[w][0];

    for (int kc = 0; kc < ATT_S; kc += 32) {
        v8f s0 = {}, s1 = {};
        {
            const __bf16* krow0 = Kr + qkBase + (size_t)(kc + l15) * ATT_D;
            const __bf16* krow1 = krow0 + 16 * ATT_D;
            U u00, u01, u10, u11;
            u00.h8[0] = *(const v8bf*)(krow0 + klo);      u00.h8[1] = *(const v8bf*)(krow0 + 16 + klo);
            u01.h8[0] = *(const v8bf*)(krow0 + 32 + klo); u01.h8[1] = *(const v8bf*)(krow0 + 48 + klo);
            u10.h8[0] = *(const v8bf*)(krow1 + klo);      u10.h8[1] = *(const v8bf*)(krow1 + 16 + klo);
            u11.h8[0] = *(const v8bf*)(krow1 + 32 + klo); u11.h8[1] = *(const v8bf*)(krow1 + 48 + klo);
            v16bf b00 = u00.v, b01 = u01.v, b10 = u10.v, b11 = u11.v;
            s0 = __builtin_amdgcn_wmma_f32_16x16x32_bf16(false, aq[0], false, b00, (short)0, s0, false, false);
            s0 = __builtin_amdgcn_wmma_f32_16x16x32_bf16(false, aq[1], false, b01, (short)0, s0, false, false);
            s1 = __builtin_amdgcn_wmma_f32_16x16x32_bf16(false, aq[0], false, b10, (short)0, s1, false, false);
            s1 = __builtin_amdgcn_wmma_f32_16x16x32_bf16(false, aq[1], false, b11, (short)0, s1, false, false);
        }

        #pragma unroll
        for (int r = 0; r < 8; ++r) {
            float mx = fmaxf(s0[r], s1[r]);
            #pragma unroll
            for (int off = 8; off >= 1; off >>= 1)
                mx = fmaxf(mx, __shfl_xor(mx, off, 32));
            float m_new = fmaxf(m_i[r], mx);
            float corr  = __expf(m_i[r] - m_new);
            float p0    = __expf(s0[r] - m_new);
            float p1    = __expf(s1[r] - m_new);
            float rs    = p0 + p1;
            #pragma unroll
            for (int off = 8; off >= 1; off >>= 1)
                rs += __shfl_xor(rs, off, 32);
            l_i[r] = l_i[r] * corr + rs;
            m_i[r] = m_new;
            #pragma unroll
            for (int t = 0; t < 4; ++t) acc[t][r] *= corr;

            int row = r + lhi * 8;
            myP[row * 32 + l15]      = (__bf16)p0;
            myP[row * 32 + 16 + l15] = (__bf16)p1;
        }
        asm volatile("s_wait_dscnt 0" ::: "memory");

        v16bf ap;
        {
            U u;
            const __bf16* prow = myP + l15 * 32;
            u.h8[0] = *(const v8bf*)(prow + klo);
            u.h8[1] = *(const v8bf*)(prow + 16 + klo);
            ap = u.v;
        }

        #pragma unroll
        for (int t = 0; t < 4; ++t) {
            const __bf16* vrow = Vt + vBase + (size_t)(t * 16 + l15) * ATT_S + kc;
            U uv;
            uv.h8[0] = *(const v8bf*)(vrow + klo);
            uv.h8[1] = *(const v8bf*)(vrow + 16 + klo);
            v16bf bv = uv.v;
            acc[t] = __builtin_amdgcn_wmma_f32_16x16x32_bf16(false, ap, false, bv, (short)0, acc[t], false, false);
        }
        asm volatile("s_wait_dscnt 0" ::: "memory");
    }

    #pragma unroll
    for (int t = 0; t < 4; ++t) {
        #pragma unroll
        for (int r = 0; r < 8; ++r) {
            int q = q0 + r + lhi * 8;
            int d = t * 16 + l15;
            float v = acc[t][r] / l_i[r];
            Ctx[((size_t)(b * ATT_S + q) * ATT_H + h) * ATT_D + d] = (__bf16)v;
        }
    }
}

// ---------------------------------------------------------------------------
// host-side launch
// ---------------------------------------------------------------------------
extern "C" void kernel_launch(void* const* d_in, const int* in_sizes, int n_in,
                              void* d_out, int out_size, void* d_ws, size_t ws_size,
                              hipStream_t stream) {
    (void)in_sizes; (void)n_in; (void)out_size; (void)ws_size;
    const int B = ATT_B, S = ATT_S, E = ATT_E;
    const int M = B * S;                 // 4096

    const float* x   = (const float*)d_in[0];
    const int*   pos = (const int*)  d_in[1];
    const float* Wq  = (const float*)d_in[2];
    const float* bq  = (const float*)d_in[3];
    const float* Wk  = (const float*)d_in[4];
    const float* bk  = (const float*)d_in[5];
    const float* Wv  = (const float*)d_in[6];
    const float* bv  = (const float*)d_in[7];
    const float* Wo  = (const float*)d_in[8];
    const float* bo  = (const float*)d_in[9];

    char* p = (char*)d_ws;
    __bf16* xb  = (__bf16*)p; p += (size_t)M * E * 2;
    __bf16* WtQ = (__bf16*)p; p += (size_t)E * E * 2;
    __bf16* WtK = (__bf16*)p; p += (size_t)E * E * 2;
    __bf16* WtV = (__bf16*)p; p += (size_t)E * E * 2;
    __bf16* WtO = (__bf16*)p; p += (size_t)E * E * 2;
    float*  Qf  = (float*)p;  p += (size_t)M * E * 4;
    float*  Kf  = (float*)p;  p += (size_t)M * E * 4;
    float*  Vf  = (float*)p;  p += (size_t)M * E * 4;
    __bf16* Qr  = (__bf16*)p; p += (size_t)M * E * 2;
    __bf16* Kr  = (__bf16*)p; p += (size_t)M * E * 2;
    __bf16* Vt  = (__bf16*)p; p += (size_t)M * E * 2;
    __bf16* Ctx = xb;                    // xb dead after the V projection

    k_f32_to_bf16<<<(M * E) / 256, 256, 0, stream>>>(x, xb, M * E);

    dim3 tb(32, 8), tg(E / 32, E / 32);
    k_transpose_bf16<<<tg, tb, 0, stream>>>(Wq, WtQ, E, E);
    k_transpose_bf16<<<tg, tb, 0, stream>>>(Wk, WtK, E, E);
    k_transpose_bf16<<<tg, tb, 0, stream>>>(Wv, WtV, E, E);
    k_transpose_bf16<<<tg, tb, 0, stream>>>(Wo, WtO, E, E);

    dim3 gg(E / GK_BN, M / GK_BM);       // (8, 32)
    k_gemm_bf16_wmma<<<gg, 256, 0, stream>>>(xb, WtQ, bq, Qf, M, E, E);
    k_gemm_bf16_wmma<<<gg, 256, 0, stream>>>(xb, WtK, bk, Kf, M, E, E);
    k_gemm_bf16_wmma<<<gg, 256, 0, stream>>>(xb, WtV, bv, Vf, M, E, E);

    k_rope_pack<<<(B * ATT_H * S * 32) / 256, 256, 0, stream>>>(Qf, Kf, pos, Qr, Kr);
    k_v_transpose<<<(M * E) / 256, 256, 0, stream>>>(Vf, Vt);

    dim3 ag(S / 128, B * ATT_H);
    k_attn_fa<<<ag, 256, 0, stream>>>(Qr, Kr, Vt, Ctx);

    k_gemm_bf16_wmma<<<gg, 256, 0, stream>>>(Ctx, WtO, bo, (float*)d_out, M, E, E);
}
